// CFTLayer1d_89541478187713
// MI455X (gfx1250) — matrix operand order repaired and checked by reading, not compiled, not run on hardware
//
#include <hip/hip_runtime.h>
#include <math.h>

typedef __attribute__((ext_vector_type(16))) _Float16 v16h;
typedef __attribute__((ext_vector_type(8)))  float    v8f;
typedef __attribute__((ext_vector_type(4)))  float    f32x4;  // native vector: OK for nontemporal builtins

#define B_   8
#define C_   64
#define L_   65536
#define S_   4
#define M_   32
#define O_   64
#define G_   16384   /* L_/S_ */

// Monotone order-preserving float<->uint encoding so integer atomicMin/Max
// implement exact float min/max (deterministic, order independent).
__device__ __forceinline__ unsigned enc_f(float f) {
  unsigned u = __float_as_uint(f);
  return (u & 0x80000000u) ? ~u : (u | 0x80000000u);
}
__device__ __forceinline__ float dec_f(unsigned k) {
  return (k & 0x80000000u) ? __uint_as_float(k ^ 0x80000000u)
                           : __uint_as_float(~k);
}

// ---------------------------------------------------------------- kernel 1
__global__ void k_init(unsigned* __restrict__ minkey, unsigned* __restrict__ maxkey) {
  int t = threadIdx.x;
  if (t < S_) { minkey[t] = 0xFFFFFFFFu; maxkey[t] = 0u; }
}

// ---------------------------------------------------------------- kernel 2
// Per-segment global min/max. Each block covers 4096 contiguous elements
// (aligned, so it never straddles a segment boundary: G_=16384 is a multiple).
// Loads use the default RT temporal hint so x is left resident in the 192 MB
// L2 for the second pass (x = 134 MB < L2).
__global__ void k_minmax(const float* __restrict__ x,
                         unsigned* __restrict__ minkey,
                         unsigned* __restrict__ maxkey) {
  const int tid = threadIdx.x;
  const long long base4 = (long long)blockIdx.x * 1024;  // in f32x4 units
  const f32x4* __restrict__ x4 = (const f32x4*)x;
  float mn =  3.4e38f, mx = -3.4e38f;
#pragma unroll
  for (int j = 0; j < 4; ++j) {
    f32x4 v = x4[base4 + tid + 256 * j];
    mn = fminf(mn, fminf(fminf(v[0], v[1]), fminf(v[2], v[3])));
    mx = fmaxf(mx, fmaxf(fmaxf(v[0], v[1]), fmaxf(v[2], v[3])));
  }
#pragma unroll
  for (int off = 16; off; off >>= 1) {
    mn = fminf(mn, __shfl_xor(mn, off, 32));
    mx = fmaxf(mx, __shfl_xor(mx, off, 32));
  }
  __shared__ float smn[8], smx[8];
  const int wave = tid >> 5;
  if ((tid & 31) == 0) { smn[wave] = mn; smx[wave] = mx; }
  __syncthreads();
  if (tid == 0) {
    float bmn = smn[0], bmx = smx[0];
#pragma unroll
    for (int w = 1; w < 8; ++w) { bmn = fminf(bmn, smn[w]); bmx = fmaxf(bmx, smx[w]); }
    const int s = (blockIdx.x & 15) >> 2;   // (base % L_) / G_
    atomicMin(&minkey[s], enc_f(bmn));
    atomicMax(&maxkey[s], enc_f(bmx));
  }
}

// ---------------------------------------------------------------- kernel 3
// One block per (b,c,s): 32 Chebyshev coefficients c_m = mean_G(xn * T_m(xn)).
// Second (and last) read of x: non-temporal loads — an L2 hit still hits, but
// the lines are not retained (x is dead after this pass).
__global__ void k_coeffs(const float* __restrict__ x,
                         const unsigned* __restrict__ minkey,
                         const unsigned* __restrict__ maxkey,
                         float* __restrict__ coeffs) {
  const int tid = threadIdx.x;
  const int idx = blockIdx.x;       // = (b*C_ + c)*S_ + s
  const int s   = idx & 3;
  const long long bc = idx >> 2;
  const float xmin = dec_f(minkey[s]);
  const float xmax = dec_f(maxkey[s]);
  const float scale = 2.0f / (xmax - xmin);
  const float bias  = -xmin * scale - 1.0f;

  const f32x4* __restrict__ x4 =
      (const f32x4*)(x + bc * (long long)L_ + (long long)s * G_);

  float acc[M_];
#pragma unroll
  for (int m = 0; m < M_; ++m) acc[m] = 0.f;

  for (int j = 0; j < 16; ++j) {           // 256 thr * 4 * 16 = 16384 = G_
    f32x4 v = __builtin_nontemporal_load(&x4[tid + 256 * j]);
#pragma unroll
    for (int q = 0; q < 4; ++q) {
      const float xn = fmaf(v[q], scale, bias);
      float tpp = 1.0f, tp = xn;
      acc[0] += xn;           // xn * T0
      acc[1] += xn * xn;      // xn * T1
#pragma unroll
      for (int m = 2; m < M_; ++m) {
        const float tn = fmaf(2.0f * xn, tp, -tpp);   // Chebyshev recurrence
        acc[m] = fmaf(xn, tn, acc[m]);
        tpp = tp; tp = tn;
      }
    }
  }
#pragma unroll
  for (int m = 0; m < M_; ++m)
#pragma unroll
    for (int off = 16; off; off >>= 1)
      acc[m] += __shfl_xor(acc[m], off, 32);

  __shared__ float sred[8 * M_];
  const int wave = tid >> 5;
  if ((tid & 31) == 0) {
#pragma unroll
    for (int m = 0; m < M_; ++m) sred[wave * M_ + m] = acc[m];
  }
  __syncthreads();
  if (tid < M_) {
    float sum = 0.f;
#pragma unroll
    for (int w = 0; w < 8; ++w) sum += sred[w * M_ + tid];
    coeffs[idx * M_ + tid] = sum * (1.0f / (float)G_);
  }
}

// ---------------------------------------------------------------- kernel 4
// One wave32 per (s,c): y[b,o] = sum_i coeffs[b,c,s,i] * W[s,i,c,o] as a
// 16x16x32 f16 WMMA (8 real batch rows padded to 16; K = 32 modes exactly).
// Four o-tiles chained through the f32 accumulator == partial sum over o,
// finished with a width-16 lane reduction; then tanh(mean_o).
__global__ void k_einsum_wmma(const float* __restrict__ coeffs,
                              const float* __restrict__ W,
                              float* __restrict__ r) {
  const int lane    = threadIdx.x & 31;
  const int wglobal = blockIdx.x * (blockDim.x >> 5) + (threadIdx.x >> 5); // 0..255
  const int s = wglobal & 3;
  const int c = wglobal >> 2;

  const int row   = lane & 15;   // A row (batch b; rows 8..15 are zero pad)
  const int khalf = lane >> 4;   // lanes 16..31 hold the other K half

  // A: 16x32 f16 (16-bit A layout: lane=M row, VGPR v -> K pairs)
  v16h A;
#pragma unroll
  for (int v = 0; v < 8; ++v) {
    const int k0 = ((v >= 4) ? 16 : 0) + ((v & 3) * 2) + khalf * 8;
    float f0 = 0.f, f1 = 0.f;
    if (row < B_) {
      const float* cp = coeffs + (((row * C_ + c) * S_ + s) * M_);
      f0 = cp[k0]; f1 = cp[k0 + 1];
    }
    A[2 * v]     = (_Float16)f0;
    A[2 * v + 1] = (_Float16)f1;
  }

  v8f acc = {};  // f32 accumulator; chaining the 4 o-tiles sums over o-tiles
#pragma unroll
  for (int t = 0; t < 4; ++t) {
    // B tile: 32x16 f16 (lane = N column; same K layout as A: khalf selects K 8-15/24-31 vs 0-7/16-23)
    v16h Bv;
    const int n = row;
#pragma unroll
    for (int v = 0; v < 8; ++v) {
      const int k0 = ((v >= 4) ? 16 : 0) + ((v & 3) * 2) + khalf * 8;
      const float g0 = W[(((s * M_ + k0)     * C_ + c) * O_) + t * 16 + n];
      const float g1 = W[(((s * M_ + k0 + 1) * C_ + c) * O_) + t * 16 + n];
      Bv[2 * v]     = (_Float16)g0;
      Bv[2 * v + 1] = (_Float16)g1;
    }
    acc = __builtin_amdgcn_wmma_f32_16x16x32_f16(
        /*neg_a=*/false, A, /*neg_b=*/false, Bv,
        /*c_mod=*/(short)0, acc, /*reuse_a=*/false, /*reuse_b=*/false);
  }

  // D layout: VGPR i of lanes 0-15 = (M=i, N=lane). Reduce over N (16 lanes).
  union { v8f v; float f[8]; } u;
  u.v = acc;
#pragma unroll
  for (int i = 0; i < 8; ++i) {
    float val = u.f[i];
#pragma unroll
    for (int off = 8; off; off >>= 1)
      val += __shfl_xor(val, off, 16);
    u.f[i] = val;
  }
  if (lane < B_) {
    const float ysum = u.f[lane];                 // row b = lane, sum over 64 o
    r[(lane * C_ + c) * S_ + s] = tanhf(ysum * (1.0f / (float)O_));
  }
}

// ---------------------------------------------------------------- kernel 5
// Broadcast r[b,c,s] over the segment with 128-bit non-temporal stores
// (streaming output: 134 MB that must not thrash L2).
__global__ void k_broadcast(const float* __restrict__ r, float* __restrict__ out) {
  const int idx = blockIdx.x;              // bc*4 + s
  const int tid = threadIdx.x;
  const float val = r[idx];
  const int s = idx & 3;
  const long long bc = idx >> 2;
  f32x4* __restrict__ out4 = (f32x4*)(out + bc * (long long)L_ + (long long)s * G_);
  f32x4 v4;
  v4[0] = val; v4[1] = val; v4[2] = val; v4[3] = val;
#pragma unroll
  for (int j = 0; j < 16; ++j)
    __builtin_nontemporal_store(v4, &out4[tid + 256 * j]);
}

// ----------------------------------------------------------------
extern "C" void kernel_launch(void* const* d_in, const int* in_sizes, int n_in,
                              void* d_out, int out_size, void* d_ws, size_t ws_size,
                              hipStream_t stream) {
  (void)in_sizes; (void)n_in; (void)out_size; (void)ws_size;
  const float* x = (const float*)d_in[0];                 // [B,C,L]
  const float* W = (const float*)d_in[1];                 // [S,M,C,O]
  float* out = (float*)d_out;                             // [B,C,L]

  unsigned* minkey = (unsigned*)d_ws;                     // 4 u32
  unsigned* maxkey = minkey + 4;                          // 4 u32
  float* coeffs = (float*)((char*)d_ws + 64);             // B*C*S*M = 65536 f32
  float* rbuf   = (float*)((char*)d_ws + 64 + (size_t)B_ * C_ * S_ * M_ * 4); // 2048 f32

  k_init      <<<1, 64, 0, stream>>>(minkey, maxkey);
  k_minmax    <<<(B_ * C_ * L_) / 4096, 256, 0, stream>>>(x, minkey, maxkey);
  k_coeffs    <<<B_ * C_ * S_, 256, 0, stream>>>(x, minkey, maxkey, coeffs);
  k_einsum_wmma<<<32, 256, 0, stream>>>(coeffs, W, rbuf);
  k_broadcast <<<B_ * C_ * S_, 256, 0, stream>>>(rbuf, out);
}
